// LinearAttention_1391569404050
// MI455X (gfx1250) — matrix operand order repaired and checked by reading, not compiled, not run on hardware
//
#include <hip/hip_runtime.h>
#include <hip/hip_bf16.h>
#include <math.h>

typedef __attribute__((ext_vector_type(16))) _Float16 v16h;
typedef __attribute__((ext_vector_type(8)))  float    v8f;

#define B_BATCH  32
#define C_DIM    256
#define NPIX     4096
#define HIDDEN   128
#define QKV_ROWS 384
#define HEADS    4

// ---------------- WMMA fragment helpers (CDNA5 16x16x32 f16 layouts) ----------------

// A-matrix 16x32 f16, this code's k order: lanes 0-15 M=0-15 hold K {0..15}; lanes 16-31 hold K {16..31}
__device__ __forceinline__ v16h load_a_frag(const _Float16* __restrict__ src, int ld) {
  const int lane = threadIdx.x & 31;
  const int m  = lane & 15;
  const int ko = (lane >> 4) << 4;            // 0 or 16
  const _Float16* row = src + (size_t)m * ld;
  v16h a;
#pragma unroll
  for (int j = 0; j < 8; ++j) a[j]     = row[ko + j];      // 16B chunk
#pragma unroll
  for (int j = 0; j < 8; ++j) a[8 + j] = row[ko + 8 + j];  // 16B chunk
  return a;
}

// B-matrix 32x16 where memory holds the transpose: elem(k,n) = src[n*ld + k]
// (per-lane reads are 32 contiguous bytes -> merges to 2x b128)
__device__ __forceinline__ v16h load_b_frag_cm(const _Float16* __restrict__ src, int ld) {
  const int lane = threadIdx.x & 31;
  const int n  = lane & 15;
  const int ko = (lane >> 4) << 4;
  const _Float16* col = src + (size_t)n * ld + ko;
  v16h b;
#pragma unroll
  for (int j = 0; j < 16; ++j) b[j] = col[j];
  return b;
}

// Fragment-packed LDS read: lane's 16 halves contiguous
__device__ __forceinline__ v16h load_frag_packed(const _Float16* __restrict__ base, int frag) {
  const int lane = threadIdx.x & 31;
  return *(const v16h*)(base + ((size_t)(frag * 32 + lane) << 4));
}

// C/D f32 16x16: lane n = lane&15; VGPR r -> M = r + 8*(lane>=16)
__device__ __forceinline__ void store_d_f16(_Float16* __restrict__ dst, int ld, v8f d) {
  const int lane = threadIdx.x & 31;
  const int n  = lane & 15;
  const int mb = (lane >> 4) << 3;
#pragma unroll
  for (int r = 0; r < 8; ++r) dst[(size_t)(mb + r) * ld + n] = (_Float16)d[r];
}

__device__ __forceinline__ v8f wmma_f16(v16h a, v16h b, v8f c) {
  return __builtin_amdgcn_wmma_f32_16x16x32_f16(false, a, false, b, (short)0, c, false, false);
}

// ---------------- Kernel 0: weight f16 conversion ----------------

__global__ void convert_weights_kernel(const float* __restrict__ Wqkv,
                                       const float* __restrict__ Wout,
                                       _Float16* __restrict__ wq,
                                       _Float16* __restrict__ wo) {
  int i = blockIdx.x * blockDim.x + threadIdx.x;
  if (i < QKV_ROWS * C_DIM) wq[i] = (_Float16)Wqkv[i];
  int j = i - QKV_ROWS * C_DIM;
  if (j >= 0 && j < C_DIM * HIDDEN) wo[j] = (_Float16)Wout[j];
}

// ---------------- Kernel 1: channel RMS-norm + g1 fold + f16 cast ----------------

__global__ void __launch_bounds__(256) prep_norm_kernel(const float* __restrict__ x,
                                                        const float* __restrict__ g1,
                                                        _Float16* __restrict__ xh) {
  const int idx = blockIdx.x * 256 + threadIdx.x;      // over B*NPIX
  const int b = idx >> 12;
  const int n = idx & (NPIX - 1);
  const float* xp = x + ((size_t)b * C_DIM) * NPIX + n;
  float ss = 0.f;
#pragma unroll 4
  for (int c = 0; c < C_DIM; ++c) { float v = xp[(size_t)c * NPIX]; ss += v * v; }
  const float s = 16.0f / fmaxf(sqrtf(ss), 1e-12f);    // sqrt(256) / ||x||
  _Float16* xo = xh + ((size_t)b * C_DIM) * NPIX + n;
#pragma unroll 4
  for (int c = 0; c < C_DIM; ++c)
    xo[(size_t)c * NPIX] = (_Float16)(xp[(size_t)c * NPIX] * s * g1[c]);
}

// ---------------- Kernel 2: QKV GEMM  [384x256] x [256x4096] -> f16 ----------------
// Phase 1: LDS B tile fragment-packed: frag (kk,ns) -> smem[((kk*4+ns)*32+lane)*16 + j]
// Phase 2: D tiles staged in LDS row-major, written out as 128B rows.

__global__ void __launch_bounds__(256) qkv_gemm_kernel(const _Float16* __restrict__ wq,
                                                       const _Float16* __restrict__ xh,
                                                       _Float16* __restrict__ qkvh) {
  __shared__ __align__(32) _Float16 smem[QKV_ROWS * 64];   // 48 KB (B tile uses first 32 KB)
  const int b  = blockIdx.x;
  const int n0 = blockIdx.y * 64;
  const int tid = threadIdx.x;
  {
    const uint4* src = (const uint4*)(xh + ((size_t)b * C_DIM) * NPIX + n0);
#pragma unroll
    for (int i = 0; i < 8; ++i) {                      // 2048 uint4 total
      const int e = tid + i * 256;
      const int k = e >> 3, c = e & 7;                 // k row, halves n = 8c..8c+7
      uint4 d = src[(size_t)k * (NPIX / 8) + c];
      const _Float16* hv = (const _Float16*)&d;
      const int kk = k >> 5;
      const int jj = k & 15;
      const int hi = (k >> 4) & 1;
#pragma unroll
      for (int t = 0; t < 8; ++t) {
        const int n  = c * 8 + t;
        const int fr = kk * 4 + (n >> 4);
        const int ln = (n & 15) + hi * 16;
        smem[((fr * 32 + ln) << 4) + jj] = hv[t];
      }
    }
  }
  __syncthreads();
  const int w = tid >> 5;
  v8f acc[3][4] = {};
#pragma unroll
  for (int kk = 0; kk < 8; ++kk) {                     // K = 256, step 32
    v16h bf[4];
#pragma unroll
    for (int ns = 0; ns < 4; ++ns) bf[ns] = load_frag_packed(smem, kk * 4 + ns);
#pragma unroll
    for (int mi = 0; mi < 3; ++mi) {
      const int mt = w + mi * 8;                       // 24 M-tiles over 8 waves
      v16h a = load_a_frag(wq + (size_t)(mt * 16) * C_DIM + kk * 32, C_DIM);
#pragma unroll
      for (int ns = 0; ns < 4; ++ns) acc[mi][ns] = wmma_f16(a, bf[ns], acc[mi][ns]);
    }
  }
  __syncthreads();                                     // done reading B tile
#pragma unroll
  for (int mi = 0; mi < 3; ++mi) {                     // stage D in LDS row-major [384][64]
    const int mt = w + mi * 8;
#pragma unroll
    for (int ns = 0; ns < 4; ++ns)
      store_d_f16(smem + (size_t)(mt * 16) * 64 + ns * 16, 64, acc[mi][ns]);
  }
  __syncthreads();
  {
    uint4* dst = (uint4*)(qkvh + ((size_t)b * QKV_ROWS) * NPIX + n0);
    const uint4* ls = (const uint4*)smem;
#pragma unroll
    for (int i = 0; i < 12; ++i) {                     // 3072 uint4
      const int e = tid + i * 256;
      const int row = e >> 3, c = e & 7;               // 8 uint4 per 64-half row
      dst[(size_t)row * (NPIX / 8) + c] = ls[e];
    }
  }
}

// ---------------- Kernel 3: per-(b,h) softmaxes + linear attention ----------------

__global__ void __launch_bounds__(256) attention_kernel(_Float16* __restrict__ qkvh,
                                                        _Float16* __restrict__ qT,
                                                        _Float16* __restrict__ attnh) {
  const int b = blockIdx.x >> 2;
  const int h = blockIdx.x & 3;
  const int tid  = threadIdx.x;
  const int w    = tid >> 5;
  const int lane = tid & 31;
  _Float16* q = qkvh + ((size_t)b * QKV_ROWS + h * 32) * NPIX;
  _Float16* k = qkvh + ((size_t)b * QKV_ROWS + 128 + h * 32) * NPIX;
  _Float16* v = qkvh + ((size_t)b * QKV_ROWS + 256 + h * 32) * NPIX;
  _Float16* qt = qT + ((size_t)(b * HEADS + h)) * NPIX * 32;   // [n][d] layout

  // k: softmax over tokens (rows of 4096); one wave per row, in place, shfl reductions
  for (int d = w; d < 32; d += 8) {
    _Float16* row = k + (size_t)d * NPIX;
    float m = -3.0e38f;
    for (int i = lane; i < NPIX; i += 32) m = fmaxf(m, (float)row[i]);
#pragma unroll
    for (int off = 16; off > 0; off >>= 1) m = fmaxf(m, __shfl_xor(m, off, 32));
    float s = 0.f;
    for (int i = lane; i < NPIX; i += 32) s += __expf((float)row[i] - m);
#pragma unroll
    for (int off = 16; off > 0; off >>= 1) s += __shfl_xor(s, off, 32);
    const float inv = 1.0f / s;
    for (int i = lane; i < NPIX; i += 32)
      row[i] = (_Float16)(__expf((float)row[i] - m) * inv);
  }
  // q: softmax over head-dim (columns of 32) * DIM_HEAD^-0.5, written TRANSPOSED to qt
  const float SCALE = 0.17677669529663689f;
  for (int n = tid; n < NPIX; n += 256) {
    float qv[32];
    float m = -3.0e38f;
#pragma unroll
    for (int d = 0; d < 32; ++d) { qv[d] = (float)q[(size_t)d * NPIX + n]; m = fmaxf(m, qv[d]); }
    float s = 0.f;
#pragma unroll
    for (int d = 0; d < 32; ++d) { qv[d] = __expf(qv[d] - m); s += qv[d]; }
    const float inv = SCALE / s;
    v16h lo, hi;
#pragma unroll
    for (int d = 0; d < 16; ++d) { lo[d] = (_Float16)(qv[d] * inv); hi[d] = (_Float16)(qv[16 + d] * inv); }
    *(v16h*)(qt + (size_t)n * 32)      = lo;
    *(v16h*)(qt + (size_t)n * 32 + 16) = hi;
  }
  __syncthreads();

  // context[d,e] = sum_n k[d,n]*v[e,n] : 32x32, K=4096 split across 8 waves
  __shared__ float part[8][4][256];                    // 32 KB partials
  __shared__ __align__(32) _Float16 ctxT[32 * 32];     // context^T in f16
  {
    v8f acc[2][2] = {};
    const int kbase0 = w * (NPIX / 8);
    for (int kk = 0; kk < NPIX / 8; kk += 32) {
      const int kb = kbase0 + kk;
      v16h a0 = load_a_frag(k + kb, NPIX);
      v16h a1 = load_a_frag(k + (size_t)16 * NPIX + kb, NPIX);
      v16h b0 = load_b_frag_cm(v + kb, NPIX);
      v16h b1 = load_b_frag_cm(v + (size_t)16 * NPIX + kb, NPIX);
      acc[0][0] = wmma_f16(a0, b0, acc[0][0]);
      acc[0][1] = wmma_f16(a0, b1, acc[0][1]);
      acc[1][0] = wmma_f16(a1, b0, acc[1][0]);
      acc[1][1] = wmma_f16(a1, b1, acc[1][1]);
    }
    const int n = lane & 15, mb = (lane >> 4) << 3;
#pragma unroll
    for (int t = 0; t < 4; ++t) {
      v8f d = acc[t >> 1][t & 1];
#pragma unroll
      for (int r = 0; r < 8; ++r) part[w][t][(mb + r) * 16 + n] = d[r];
    }
  }
  __syncthreads();
  for (int e = tid; e < 1024; e += 256) {              // reduce 8 wave partials
    const int t = e >> 8, i = e & 255;
    float s = 0.f;
#pragma unroll
    for (int ww = 0; ww < 8; ++ww) s += part[ww][t][i];
    const int mt = t >> 1, et = t & 1;
    const int d  = mt * 16 + (i >> 4);
    const int ee = et * 16 + (i & 15);
    ctxT[ee * 32 + d] = (_Float16)s;
  }
  __syncthreads();

  // out[e,n] = sum_d ctxT[e,d]*qT[n,d] : K=32 -> one WMMA per 16x16 tile
  v16h ae0 = load_a_frag(ctxT, 32);
  v16h ae1 = load_a_frag(ctxT + 16 * 32, 32);
  _Float16* o0 = attnh + ((size_t)b * HIDDEN + h * 32) * NPIX;
  const v8f z = {};
  for (int nt = w; nt < NPIX / 16; nt += 8) {
    v16h bb = load_b_frag_cm(qt + (size_t)(nt * 16) * 32, 32);   // contiguous per lane
    v8f d0 = wmma_f16(ae0, bb, z);
    v8f d1 = wmma_f16(ae1, bb, z);
    store_d_f16(o0 + nt * 16, NPIX, d0);
    store_d_f16(o0 + (size_t)16 * NPIX + nt * 16, NPIX, d1);
  }
}

// ---------------- Kernel 4: Wout GEMM + bias + RMS-norm + g2 + residual ----------------
// LDS B tile fragment-packed; f32 result staged in LDS; vectorized residual+store phase.

__global__ void __launch_bounds__(256) out_gemm_norm_kernel(
    const _Float16* __restrict__ wo, const _Float16* __restrict__ attnh,
    const float* __restrict__ bout, const float* __restrict__ g2,
    const float* __restrict__ x, float* __restrict__ out) {
  __shared__ __align__(32) _Float16 Bs[HIDDEN * 32];   // 8 KB
  __shared__ float outs[C_DIM * 32];                   // 32 KB staged result
  __shared__ float colsq[32];
  __shared__ float scl[32];
  const int b  = blockIdx.x;
  const int n0 = blockIdx.y * 32;
  const int tid = threadIdx.x;
  if (tid < 32) colsq[tid] = 0.f;
  // prefetch residual rows into cache while the GEMM runs (global_prefetch_b8)
  __builtin_prefetch(x + ((size_t)b * C_DIM + tid) * NPIX + n0, 0, 1);
  {
    const uint4* src = (const uint4*)(attnh + ((size_t)b * HIDDEN) * NPIX + n0);
#pragma unroll
    for (int i = 0; i < 2; ++i) {                      // 512 uint4
      const int e = tid + i * 256;
      const int k = e >> 2, c = e & 3;
      uint4 d = src[(size_t)k * (NPIX / 8) + c];
      const _Float16* hv = (const _Float16*)&d;
      const int kk = k >> 5;
      const int jj = k & 15;
      const int hi = (k >> 4) & 1;
#pragma unroll
      for (int t = 0; t < 8; ++t) {
        const int n  = c * 8 + t;
        const int fr = kk * 2 + (n >> 4);
        const int ln = (n & 15) + hi * 16;
        Bs[((fr * 32 + ln) << 4) + jj] = hv[t];
      }
    }
  }
  __syncthreads();
  const int w = tid >> 5;
  const int mt0 = w * 2;                               // 16 M-tiles over 8 waves
  v8f acc[2][2] = {};
#pragma unroll
  for (int kk = 0; kk < 4; ++kk) {                     // K = 128, step 32
    v16h b0 = load_frag_packed(Bs, kk * 2 + 0);
    v16h b1 = load_frag_packed(Bs, kk * 2 + 1);
    v16h a0 = load_a_frag(wo + (size_t)(mt0 * 16) * HIDDEN + kk * 32, HIDDEN);
    v16h a1 = load_a_frag(wo + (size_t)((mt0 + 1) * 16) * HIDDEN + kk * 32, HIDDEN);
    acc[0][0] = wmma_f16(a0, b0, acc[0][0]);
    acc[0][1] = wmma_f16(a0, b1, acc[0][1]);
    acc[1][0] = wmma_f16(a1, b0, acc[1][0]);
    acc[1][1] = wmma_f16(a1, b1, acc[1][1]);
  }
  // epilogue: +bias, per-column sum of squares
  const int ln = tid & 31;
  const int n  = ln & 15;
  const int mb = (ln >> 4) << 3;
  float vals[2][2][8];
  float ss[2] = {0.f, 0.f};
#pragma unroll
  for (int mi = 0; mi < 2; ++mi) {
#pragma unroll
    for (int ns = 0; ns < 2; ++ns) {
#pragma unroll
      for (int r = 0; r < 8; ++r) {
        const int c = (mt0 + mi) * 16 + mb + r;
        float f = acc[mi][ns][r] + bout[c];
        vals[mi][ns][r] = f;
        ss[ns] += f * f;
      }
    }
  }
  atomicAdd(&colsq[n], ss[0]);
  atomicAdd(&colsq[16 + n], ss[1]);
  __syncthreads();
  if (tid < 32) scl[tid] = 16.0f / fmaxf(sqrtf(colsq[tid]), 1e-12f);
  __syncthreads();
  // RMS scale + g2, staged into LDS row-major [256][32]
#pragma unroll
  for (int mi = 0; mi < 2; ++mi) {
#pragma unroll
    for (int ns = 0; ns < 2; ++ns) {
#pragma unroll
      for (int r = 0; r < 8; ++r) {
        const int c   = (mt0 + mi) * 16 + mb + r;
        const int ncl = ns * 16 + n;
        outs[c * 32 + ncl] = vals[mi][ns][r] * scl[ncl] * g2[c];
      }
    }
  }
  __syncthreads();
  // vectorized residual add + store (float4 in, float4 out)
  {
    const float4* xs = (const float4*)(x   + ((size_t)b * C_DIM) * NPIX + n0);
    float4*       os = (float4*)(out       + ((size_t)b * C_DIM) * NPIX + n0);
    const float4* ls = (const float4*)outs;
#pragma unroll
    for (int i = 0; i < 8; ++i) {                      // 2048 float4
      const int e = tid + i * 256;
      const int row = e >> 3, c = e & 7;               // 8 float4 per 32-float row
      float4 xv = xs[(size_t)row * (NPIX / 4) + c];
      float4 lv = ls[e];
      float4 r4;
      r4.x = lv.x + xv.x; r4.y = lv.y + xv.y; r4.z = lv.z + xv.z; r4.w = lv.w + xv.w;
      os[(size_t)row * (NPIX / 4) + c] = r4;
    }
  }
}

// ---------------- launch ----------------

extern "C" void kernel_launch(void* const* d_in, const int* in_sizes, int n_in,
                              void* d_out, int out_size, void* d_ws, size_t ws_size,
                              hipStream_t stream) {
  const float* x    = (const float*)d_in[0];
  const float* g1   = (const float*)d_in[1];
  const float* Wqkv = (const float*)d_in[2];
  const float* Wout = (const float*)d_in[3];
  const float* bout = (const float*)d_in[4];
  const float* g2   = (const float*)d_in[5];
  float* out = (float*)d_out;

  char* ws = (char*)d_ws;
  _Float16* xh    = (_Float16*)(ws);                                   // 64 MiB
  _Float16* qkvh  = (_Float16*)(ws + (((size_t)64)  << 20));           // 96 MiB
  _Float16* attnh = (_Float16*)(ws + (((size_t)160) << 20));           // 32 MiB
  _Float16* qT    = (_Float16*)(ws + (((size_t)192) << 20));           // 32 MiB
  _Float16* wq    = (_Float16*)(ws + (((size_t)224) << 20));           // 192 KiB
  _Float16* wo    = (_Float16*)(ws + (((size_t)224) << 20) + (256 << 10));

  convert_weights_kernel<<<(QKV_ROWS * C_DIM + C_DIM * HIDDEN + 255) / 256, 256, 0, stream>>>(
      Wqkv, Wout, wq, wo);
  prep_norm_kernel<<<(B_BATCH * NPIX) / 256, 256, 0, stream>>>(x, g1, xh);

  dim3 gq(B_BATCH, NPIX / 64);
  qkv_gemm_kernel<<<gq, 256, 0, stream>>>(wq, xh, qkvh);

  attention_kernel<<<B_BATCH * HEADS, 256, 0, stream>>>(qkvh, qT, attnh);

  dim3 go(B_BATCH, NPIX / 32);
  out_gemm_norm_kernel<<<go, 256, 0, stream>>>(wo, attnh, bout, g2, x, out);
}
